// SupConAdapProtoHardLossGroupNorm_60541859004772
// MI455X (gfx1250) — matrix operand order repaired and checked by reading, not compiled, not run on hardware
//
#include <hip/hip_runtime.h>
#include <hip/hip_bf16.h>

#define NROWS 4096
#define DDIM  512
#define NGRP  8
#define JBLKS 32   // NROWS / 128

typedef __attribute__((ext_vector_type(16))) _Float16 v16h;
typedef __attribute__((ext_vector_type(8)))  _Float16 h8v;
typedef __attribute__((ext_vector_type(8)))  float    v8f;

union Frag { v16h v; h8v h[2]; };

__device__ __forceinline__ float block_reduce_sum(float v, float* red, int t) {
  red[t] = v; __syncthreads();
  #pragma unroll
  for (int s = 128; s > 0; s >>= 1) {
    if (t < s) red[t] = red[t] + red[t + s];
    __syncthreads();
  }
  float r = red[0]; __syncthreads();
  return r;
}

// ---------------- Kernel 1: group prototypes (mean per group, L2-normalized) + counts
__global__ __launch_bounds__(256) void proto_kernel(const float* __restrict__ F,
                                                    const int* __restrict__ g,
                                                    float* __restrict__ protos,
                                                    float* __restrict__ cnt) {
  __shared__ float red[256];
  const int b = blockIdx.x;
  const int t = threadIdx.x;
  float s0 = 0.f, s1 = 0.f, c = 0.f;
  for (int j = 0; j < NROWS; ++j) {
    if (g[j] == b) {
      s0 += F[(size_t)j * DDIM + t];
      s1 += F[(size_t)j * DDIM + t + 256];
      c  += 1.f;
    }
  }
  const float m0 = s0 / (c + 1e-12f);
  const float m1 = s1 / (c + 1e-12f);
  const float nrm2 = block_reduce_sum(m0 * m0 + m1 * m1, red, t);
  const float inv = 1.f / fmaxf(sqrtf(nrm2), 1e-12f);
  protos[b * DDIM + t]       = m0 * inv;
  protos[b * DDIM + t + 256] = m1 * inv;
  if (t == 0) cnt[b] = c;
}

// ---------------- Kernel 2: fp32 -> (hi, lo) f16 split for near-fp32 WMMA
__global__ __launch_bounds__(256) void split_kernel(const float* __restrict__ F,
                                                    _Float16* __restrict__ Fhi,
                                                    _Float16* __restrict__ Flo) {
  const size_t idx = (size_t)blockIdx.x * 256 + threadIdx.x;
  const float f = F[idx];
  const _Float16 hi = (_Float16)f;
  Fhi[idx] = hi;
  Flo[idx] = (_Float16)(f - (float)hi);
}

// ---------------- Kernel 3: P[i][p] = F[i].protos[p]; Mrow[i] = ||F[i]||^2 / TEMP (row max of logits)
__global__ __launch_bounds__(256) void rowprep_kernel(const float* __restrict__ F,
                                                      const float* __restrict__ protos,
                                                      float* __restrict__ P,
                                                      float* __restrict__ Mrow) {
  __shared__ float red[256];
  const int i = blockIdx.x;
  const int t = threadIdx.x;
  const float f0 = F[(size_t)i * DDIM + t];
  const float f1 = F[(size_t)i * DDIM + t + 256];
  const float invT = 1.0f / 0.07f;
  const float nrm2 = block_reduce_sum(f0 * f0 + f1 * f1, red, t);
  if (t == 0) Mrow[i] = nrm2 * invT;
  for (int p = 0; p < NGRP; ++p) {
    const float d = block_reduce_sum(f0 * protos[p * DDIM + t] + f1 * protos[p * DDIM + t + 256], red, t);
    if (t == 0) P[i * NGRP + p] = d;
  }
}

// ---------------- Kernel 4: fused WMMA GEMM (F F^T) + masked weighted row reductions
__global__ __launch_bounds__(256) void supcon_main_kernel(const _Float16* __restrict__ Fhi,
                                                          const _Float16* __restrict__ Flo,
                                                          const int* __restrict__ g,
                                                          const int* __restrict__ c,
                                                          const int* __restrict__ s,
                                                          const float* __restrict__ P,
                                                          const float* __restrict__ Mrow,
                                                          float* __restrict__ denomPart,
                                                          float* __restrict__ possumPart) {
  __shared__ int   s_gr[128], s_cr[128], s_sr[128];
  __shared__ int   s_gc[128], s_cc[128], s_sc[128];
  __shared__ float s_M[128];
  __shared__ float s_P[128][NGRP];
  __shared__ float s_pd[4][128];
  __shared__ float s_pp[4][128];

  const int t  = threadIdx.x;
  const int i0 = blockIdx.y * 128;
  const int j0 = blockIdx.x * 128;

  if (t < 128) {
    const int r = i0 + t;
    s_gr[t] = g[r]; s_cr[t] = c[r]; s_sr[t] = s[r]; s_M[t] = Mrow[r];
    #pragma unroll
    for (int p = 0; p < NGRP; ++p) s_P[t][p] = P[r * NGRP + p];
    const int q = j0 + t;
    s_gc[t] = g[q]; s_cc[t] = c[q]; s_sc[t] = s[q];
  }
  __syncthreads();

  const int lane = t & 31;
  const int wid  = t >> 5;
  const int wm   = wid & 1;   // 2 strips of 64 rows
  const int wn   = wid >> 1;  // 4 strips of 32 cols
  const int m    = lane & 15;
  const int hb   = lane >> 4;

  const _Float16* arh[4]; const _Float16* arl[4];
  #pragma unroll
  for (int mi = 0; mi < 4; ++mi) {
    const size_t row = (size_t)(i0 + wm * 64 + mi * 16 + m) * DDIM;
    arh[mi] = Fhi + row; arl[mi] = Flo + row;
  }
  const _Float16* brh[2]; const _Float16* brl[2];
  #pragma unroll
  for (int ni = 0; ni < 2; ++ni) {
    const size_t row = (size_t)(j0 + wn * 32 + ni * 16 + m) * DDIM;
    brh[ni] = Fhi + row; brl[ni] = Flo + row;
  }

  v8f acc[4][2];
  #pragma unroll
  for (int mi = 0; mi < 4; ++mi)
    #pragma unroll
    for (int ni = 0; ni < 2; ++ni)
      #pragma unroll
      for (int e = 0; e < 8; ++e) acc[mi][ni][e] = 0.f;

  for (int kk = 0; kk < DDIM; kk += 32) {
    // 16-bit A-matrix 16x32 layout: lo lanes K0-7|16-23, hi lanes K8-15|24-31
    const int kA0 = kk + 8 * hb;
    const int kA1 = kk + 16 + 8 * hb;
    // 16-bit B-matrix 32x16 layout: lo lanes K0-7|16-23, hi lanes K8-15|24-31
    const int kB0 = kk + 8 * hb;
    const int kB1 = kk + 16 + 8 * hb;

    Frag ah[4], al[4], bh[2], bl[2];
    #pragma unroll
    for (int mi = 0; mi < 4; ++mi) {
      ah[mi].h[0] = *(const h8v*)(arh[mi] + kA0);
      ah[mi].h[1] = *(const h8v*)(arh[mi] + kA1);
    }
    #pragma unroll
    for (int ni = 0; ni < 2; ++ni) {
      bh[ni].h[0] = *(const h8v*)(brh[ni] + kB0);
      bh[ni].h[1] = *(const h8v*)(brh[ni] + kB1);
    }
    #pragma unroll
    for (int mi = 0; mi < 4; ++mi)
      #pragma unroll
      for (int ni = 0; ni < 2; ++ni)
        acc[mi][ni] = __builtin_amdgcn_wmma_f32_16x16x32_f16(
            false, ah[mi].v, false, bh[ni].v, (short)0, acc[mi][ni], false, false);

    #pragma unroll
    for (int ni = 0; ni < 2; ++ni) {
      bl[ni].h[0] = *(const h8v*)(brl[ni] + kB0);
      bl[ni].h[1] = *(const h8v*)(brl[ni] + kB1);
    }
    #pragma unroll
    for (int mi = 0; mi < 4; ++mi)
      #pragma unroll
      for (int ni = 0; ni < 2; ++ni)
        acc[mi][ni] = __builtin_amdgcn_wmma_f32_16x16x32_f16(
            false, ah[mi].v, false, bl[ni].v, (short)0, acc[mi][ni], false, false);

    #pragma unroll
    for (int mi = 0; mi < 4; ++mi) {
      al[mi].h[0] = *(const h8v*)(arl[mi] + kA0);
      al[mi].h[1] = *(const h8v*)(arl[mi] + kA1);
    }
    #pragma unroll
    for (int mi = 0; mi < 4; ++mi)
      #pragma unroll
      for (int ni = 0; ni < 2; ++ni)
        acc[mi][ni] = __builtin_amdgcn_wmma_f32_16x16x32_f16(
            false, al[mi].v, false, bh[ni].v, (short)0, acc[mi][ni], false, false);
  }

  // ---- epilogue: masked weighted per-row reductions
  const float invT = 1.0f / 0.07f;
  #pragma unroll
  for (int mi = 0; mi < 4; ++mi) {
    #pragma unroll
    for (int v = 0; v < 8; ++v) {
      const int lr = wm * 64 + mi * 16 + v + 8 * hb;  // C/D layout: row = v + 8*hb
      const int gi = s_gr[lr], ci = s_cr[lr], si = s_sr[lr];
      const float Mi = s_M[lr];
      float dsum = 0.f, psum = 0.f;
      #pragma unroll
      for (int ni = 0; ni < 2; ++ni) {
        const int lq = wn * 32 + ni * 16 + m;         // C/D layout: col = lane & 15
        if ((i0 + lr) == (j0 + lq)) continue;         // logits_mask (no diagonal)
        const float sim = acc[mi][ni][v];
        const float lg = sim * invT - Mi;
        const float ex = __expf(lg);
        const int gq = s_gc[lq];
        if (gi == gq) {                                // positive pair
          dsum += ex; psum += lg;
        } else {
          const bool classEq = (ci == s_cc[lq]);       // soft negative
          const bool sensEq  = (si == s_sc[lq]);       // hard negative if !classEq
          if (classEq || sensEq) {
            const float wcoef = classEq ? 3.0f : 5.0f;
            const float diff = sim - s_P[lr][gq];      // sim - prototype similarity
            const float w = (diff > 0.f) ? __expf(wcoef * diff) : 1.0f;
            dsum += w * ex;
          }
        }
      }
      #pragma unroll
      for (int o = 8; o > 0; o >>= 1) {               // reduce 16 lanes within each half
        dsum += __shfl_xor(dsum, o, 32);
        psum += __shfl_xor(psum, o, 32);
      }
      if (m == 0) { s_pd[wn][lr] = dsum; s_pp[wn][lr] = psum; }
    }
  }
  __syncthreads();
  if (t < 128) {  // deterministic fixed-order combine across the 4 column-strip waves
    const float d = s_pd[0][t] + s_pd[1][t] + s_pd[2][t] + s_pd[3][t];
    const float p = s_pp[0][t] + s_pp[1][t] + s_pp[2][t] + s_pp[3][t];
    denomPart[(size_t)blockIdx.x * NROWS + (i0 + t)]  = d;
    possumPart[(size_t)blockIdx.x * NROWS + (i0 + t)] = p;
  }
}

// ---------------- Kernel 5: finalize (deterministic fixed-order sums)
__global__ __launch_bounds__(256) void finalize_kernel(const float* __restrict__ denomPart,
                                                       const float* __restrict__ possumPart,
                                                       const int* __restrict__ g,
                                                       const float* __restrict__ cnt,
                                                       float* __restrict__ out) {
  __shared__ float red[256];
  __shared__ float gtot[NGRP];
  const int t = threadIdx.x;
  float ls[NGRP];
  #pragma unroll
  for (int p = 0; p < NGRP; ++p) ls[p] = 0.f;
  for (int i = t; i < NROWS; i += 256) {
    float d = 0.f, p = 0.f;
    for (int jb = 0; jb < JBLKS; ++jb) {
      d += denomPart[(size_t)jb * NROWS + i];
      p += possumPart[(size_t)jb * NROWS + i];
    }
    const int gi = g[i];
    const float cp = cnt[gi] - 1.0f;                       // positives per row
    const float mlpp = (p - cp * logf(d + 1e-12f)) / (cp + 1e-12f);
    ls[gi] += -mlpp;                                        // TEMP/BASE_TEMP == 1
  }
  for (int p = 0; p < NGRP; ++p) {
    const float v = block_reduce_sum(ls[p], red, t);
    if (t == 0) gtot[p] = v;
  }
  __syncthreads();
  if (t == 0) {
    float a = 0.f;
    for (int p = 0; p < NGRP; ++p) a += gtot[p] / fmaxf(cnt[p], 1e-12f);
    out[0] = a / (float)NGRP;
  }
}

extern "C" void kernel_launch(void* const* d_in, const int* in_sizes, int n_in,
                              void* d_out, int out_size, void* d_ws, size_t ws_size,
                              hipStream_t stream) {
  (void)in_sizes; (void)n_in; (void)out_size; (void)ws_size;
  const float* F = (const float*)d_in[0];
  const int*   g = (const int*)d_in[1];
  const int*   c = (const int*)d_in[2];
  const int*   s = (const int*)d_in[3];
  float* out = (float*)d_out;

  char* ws = (char*)d_ws;
  size_t off = 0;
  auto carve = [&](size_t bytes) -> void* {
    void* p = ws + off;
    off = (off + bytes + 255) & ~(size_t)255;
    return p;
  };
  _Float16* Fhi    = (_Float16*)carve((size_t)NROWS * DDIM * sizeof(_Float16));
  _Float16* Flo    = (_Float16*)carve((size_t)NROWS * DDIM * sizeof(_Float16));
  float* protos    = (float*)carve((size_t)NGRP * DDIM * sizeof(float));
  float* cnt       = (float*)carve((size_t)NGRP * sizeof(float));
  float* P         = (float*)carve((size_t)NROWS * NGRP * sizeof(float));
  float* Mrow      = (float*)carve((size_t)NROWS * sizeof(float));
  float* denomPart = (float*)carve((size_t)JBLKS * NROWS * sizeof(float));
  float* possumPart= (float*)carve((size_t)JBLKS * NROWS * sizeof(float));

  proto_kernel<<<NGRP, 256, 0, stream>>>(F, g, protos, cnt);
  split_kernel<<<(NROWS * DDIM) / 256, 256, 0, stream>>>(F, Fhi, Flo);
  rowprep_kernel<<<NROWS, 256, 0, stream>>>(F, protos, P, Mrow);
  supcon_main_kernel<<<dim3(JBLKS, JBLKS), 256, 0, stream>>>(Fhi, Flo, g, c, s, P, Mrow,
                                                             denomPart, possumPart);
  finalize_kernel<<<1, 256, 0, stream>>>(denomPart, possumPart, g, cnt, out);
}
